// UnifiedGNN_44057774523060
// MI455X (gfx1250) — matrix operand order repaired and checked by reading, not compiled, not run on hardware
//
#include <hip/hip_runtime.h>
#include <math.h>

// ---------------- problem constants (from reference) ----------------
#define NQ 10000
#define NT 50000
#define NA 50000
#define E_GR 50000
#define E_LT 50000
#define E_SIM 80000
#define E_COMP 100000
#define E_GEN 50000
#define HID 512
#define NHD 4
#define HC 128
#define DOTD 256
#define NL 2
#define NEG_SLOPE 0.2f

#define FLAG_BIAS 1
#define FLAG_RELU 2
#define FLAG_ACC  4

typedef __attribute__((ext_vector_type(16))) _Float16 v16h;
typedef __attribute__((ext_vector_type(8)))  _Float16 v8h;
typedef __attribute__((ext_vector_type(4)))  _Float16 v4h;
typedef __attribute__((ext_vector_type(8)))  float    v8f;

// LDS row strides (halfwords): +8 pad keeps 16B alignment and spreads banks
#define LDA_S 40   // A tile row stride (32 + 8)
#define LDB_S 40   // B tile row stride (32 + 8)

// ---------------- WMMA GEMM: C[M,N] = gather(A)[M,K] * B[K,N] (+bias)(+C)(relu) -------
// Block = 256 thr = 8 waves; tile 128x64; K-step 32.
// A staged row-major f16 in LDS; B staged TRANSPOSED f16 in LDS so that every
// WMMA fragment is two contiguous ds_load_b128 reads (ISA 7.12.2 layouts).
// OOB A rows are clamped (not zeroed): their products only reach output rows
// >= M which the epilogue never stores -> staging is branchless.
__global__ __launch_bounds__(256) void gemm_wmma_f16(
    const float* __restrict__ A, const float* __restrict__ B,
    const float* __restrict__ bias, float* __restrict__ C,
    const int* __restrict__ rowidx, int M, int N, int K, int flags)
{
  __shared__ _Float16 lA[128 * LDA_S];   // 10240 B
  __shared__ _Float16 lB[64  * LDB_S];   //  5120 B

  const int t    = threadIdx.x;
  const int lane = t & 31;
  const int wave = t >> 5;
  const int rowBase = blockIdx.y * 128;
  const int colBase = blockIdx.x * 64;
  const int mn = lane & 15;      // A-row / B-col / D-col within 16x16 tile
  const int g  = lane >> 4;      // lane half

  // ---- precompute per-thread A staging sources (4 x float4 = 16 floats) ----
  const float* Ap[4];
  int lofsA[4];
  #pragma unroll
  for (int i = 0; i < 4; ++i) {
    int idx = (i * 1024 + t * 4);          // element within 128x32 tile
    int row = idx >> 5, k = idx & 31;
    int ar  = rowBase + row;
    if (ar > M - 1) ar = M - 1;            // clamp: OOB rows never stored
    int r = rowidx ? rowidx[ar] : ar;
    Ap[i] = A + (size_t)r * K + k;
    lofsA[i] = row * LDA_S + k;
  }
  // ---- precompute per-thread B staging coords (8 elements of 32x64 tile) ----
  const float* Bp[8];
  int lofsB[8];
  #pragma unroll
  for (int i = 0; i < 8; ++i) {
    int idx = i * 256 + t;                 // 0..2047
    int n = idx & 63, k = idx >> 6;
    Bp[i] = B + (size_t)k * N + (colBase + n);
    lofsB[i] = n * LDB_S + k;
  }

  v8f acc[4];
  #pragma unroll
  for (int s = 0; s < 4; ++s) acc[s] = (v8f){0.f,0.f,0.f,0.f,0.f,0.f,0.f,0.f};

  for (int k0 = 0; k0 < K; k0 += 32) {
    __syncthreads();               // previous fragments consumed
    // ---- stage: batch all global loads first, then convert + LDS store ----
    float4 av[4];
    #pragma unroll
    for (int i = 0; i < 4; ++i) av[i] = *(const float4*)(Ap[i] + k0);
    float bv[8];
    #pragma unroll
    for (int i = 0; i < 8; ++i) bv[i] = Bp[i][(size_t)k0 * N];
    #pragma unroll
    for (int i = 0; i < 4; ++i) {
      v4h h;
      h[0] = (_Float16)av[i].x; h[1] = (_Float16)av[i].y;
      h[2] = (_Float16)av[i].z; h[3] = (_Float16)av[i].w;
      *(v4h*)&lA[lofsA[i]] = h;
    }
    #pragma unroll
    for (int i = 0; i < 8; ++i) lB[lofsB[i]] = (_Float16)bv[i];
    __syncthreads();

    // ---- fragments: 2x ds_load_b128 each; preload all B then 4 WMMAs ----
    int arow = wave * 16 + mn;
    v8h alo = *(const v8h*)&lA[arow * LDA_S + 8 * g];
    v8h ahi = *(const v8h*)&lA[arow * LDA_S + 16 + 8 * g];
    v16h af = __builtin_shufflevector(alo, ahi,
        0,1,2,3,4,5,6,7,8,9,10,11,12,13,14,15);
    v16h bf[4];
    #pragma unroll
    for (int s = 0; s < 4; ++s) {
      int bn = 16 * s + mn;
      v8h blo = *(const v8h*)&lB[bn * LDB_S + 8 * g];
      v8h bhi = *(const v8h*)&lB[bn * LDB_S + 16 + 8 * g];
      bf[s] = __builtin_shufflevector(blo, bhi,
          0,1,2,3,4,5,6,7,8,9,10,11,12,13,14,15);
    }
    #pragma unroll
    for (int s = 0; s < 4; ++s)
      acc[s] = __builtin_amdgcn_wmma_f32_16x16x32_f16(
          false, af, false, bf[s], (short)0, acc[s], false, false);
  }

  // ---- epilogue: D layout VGPR r -> M = r + 8g, N = lane&15 ----
  #pragma unroll
  for (int s = 0; s < 4; ++s) {
    int col = colBase + 16 * s + mn;
    #pragma unroll
    for (int r = 0; r < 8; ++r) {
      int row = rowBase + wave * 16 + 8 * g + r;
      if (row < M) {
        size_t o = (size_t)row * N + col;
        float v = acc[s][r];
        if (flags & FLAG_ACC)  v += C[o];
        if (flags & FLAG_BIAS) v += bias[col];
        if (flags & FLAG_RELU) v = fmaxf(v, 0.f);
        C[o] = v;
      }
    }
  }
}

// ---------------- utility kernels ----------------
__global__ void k_fill(float* __restrict__ p, size_t n, float v) {
  size_t i = (size_t)blockIdx.x * blockDim.x + threadIdx.x;
  if (i < n) p[i] = v;
}

__global__ void k_acc_init(float* __restrict__ acc, const float* __restrict__ b0,
                           const float* __restrict__ b1, const float* __restrict__ b2,
                           size_t total) {
  size_t i = (size_t)blockIdx.x * blockDim.x + threadIdx.x;
  if (i >= total) return;
  int c = (int)(i & (HID - 1));
  float v = b0[c];
  if (b1) v += b1[c];
  if (b2) v += b2[c];
  acc[i] = v;
}

__global__ void k_finalize(float* __restrict__ h, const float* __restrict__ acc,
                           size_t total, float inv) {
  size_t i = (size_t)blockIdx.x * blockDim.x + threadIdx.x;
  if (i >= total) return;
  h[i] = fmaxf(acc[i] * inv, 0.f);
}

__device__ __forceinline__ void atomicMaxFloat(float* addr, float val) {
  // monotonic int/uint mapping trick; addr initialized to -inf
  if (val >= 0.f) atomicMax((int*)addr, __float_as_int(val));
  else            atomicMin((unsigned int*)addr, __float_as_uint(val));
}

// ---------------- GATv2 edge kernels (one wave32 per edge; 16 channels/lane) ----------
__global__ __launch_bounds__(256) void k_edge_logits(
    const int* __restrict__ src, const int* __restrict__ dst, int E, int loop_mode,
    const float* __restrict__ xl, const float* __restrict__ xr,
    const float* __restrict__ att, float* __restrict__ logits,
    float* __restrict__ lmax, int loff)
{
  int wave = threadIdx.x >> 5, lane = threadIdx.x & 31;
  int e = blockIdx.x * 8 + wave;
  if (e >= E) return;
  int s = loop_mode ? e : src[e];
  int d = loop_mode ? e : dst[e];
  const float4* x4l = (const float4*)(xl + (size_t)s * HID);
  const float4* x4r = (const float4*)(xr + (size_t)d * HID);
  const float4* a4  = (const float4*)att;     // att flat [4*128] == channel index
  float part = 0.f;
  #pragma unroll
  for (int j = 0; j < 4; ++j) {
    int idx = lane * 4 + j;                   // lanes [8h,8h+8) cover head h
    float4 a = x4l[idx], b = x4r[idx], w = a4[idx];
    float v;
    v = a.x + b.x; v = (v > 0.f) ? v : NEG_SLOPE * v; part += w.x * v;
    v = a.y + b.y; v = (v > 0.f) ? v : NEG_SLOPE * v; part += w.y * v;
    v = a.z + b.z; v = (v > 0.f) ? v : NEG_SLOPE * v; part += w.z * v;
    v = a.w + b.w; v = (v > 0.f) ? v : NEG_SLOPE * v; part += w.w * v;
  }
  part += __shfl_xor(part, 1, 32);
  part += __shfl_xor(part, 2, 32);
  part += __shfl_xor(part, 4, 32);
  if ((lane & 7) == 0) {
    int h = lane >> 3;
    logits[(size_t)(loff + e) * NHD + h] = part;
    atomicMaxFloat(&lmax[(size_t)d * NHD + h], part);
  }
}

__global__ void k_edge_exp(const int* __restrict__ dst, int E, int loop_mode,
                           const float* __restrict__ logits, int loff,
                           const float* __restrict__ lmax,
                           float* __restrict__ ex, float* __restrict__ denom)
{
  int i = blockIdx.x * blockDim.x + threadIdx.x;
  if (i >= E * NHD) return;
  int e = i >> 2, h = i & 3;
  int d = loop_mode ? e : dst[e];
  float lm = lmax[(size_t)d * NHD + h];
  if (lm < -3.0e38f) lm = 0.f;        // isfinite() fix from reference
  float v = __expf(logits[(size_t)(loff + e) * NHD + h] - lm);
  ex[(size_t)(loff + e) * NHD + h] = v;
  atomicAdd(&denom[(size_t)d * NHD + h], v);
}

__global__ __launch_bounds__(256) void k_edge_scatter(
    const int* __restrict__ src, const int* __restrict__ dst, int E, int loop_mode,
    const float* __restrict__ xl, const float* __restrict__ ex, int loff,
    const float* __restrict__ denom, float* __restrict__ acc)
{
  int wave = threadIdx.x >> 5, lane = threadIdx.x & 31;
  int e = blockIdx.x * 8 + wave;
  if (e >= E) return;
  int s = loop_mode ? e : src[e];
  int d = loop_mode ? e : dst[e];
  int h = lane >> 3;
  float alpha = ex[(size_t)(loff + e) * NHD + h]
              / (denom[(size_t)d * NHD + h] + 1e-16f);
  const float4* x4 = (const float4*)(xl + (size_t)s * HID);
  float* ad = acc + (size_t)d * HID;
  #pragma unroll
  for (int j = 0; j < 4; ++j) {
    int idx = lane * 4 + j;
    float4 v = x4[idx];
    int c = idx * 4;
    atomicAdd(&ad[c + 0], v.x * alpha);
    atomicAdd(&ad[c + 1], v.y * alpha);
    atomicAdd(&ad[c + 2], v.z * alpha);
    atomicAdd(&ad[c + 3], v.w * alpha);
  }
}

// ---------------- head kernels ----------------
__global__ void k_afp1(const float* __restrict__ af, const float* __restrict__ W1,
                       const float* __restrict__ b1, float* __restrict__ out, int E)
{
  int i = blockIdx.x * blockDim.x + threadIdx.x;
  if (i >= E * 64) return;
  int e = i >> 6, j = i & 63;
  float s = b1[j];
  #pragma unroll
  for (int k = 0; k < 6; ++k) s += af[(size_t)e * 6 + k] * W1[k * 64 + j];
  out[i] = fmaxf(s, 0.f);
}

__global__ void k_bc(const float* __restrict__ b2, const float* __restrict__ W3,
                     const float* __restrict__ rhb, float* __restrict__ bc)
{
  int j = threadIdx.x;  // 256 threads
  float s = rhb[j];
  for (int k = 0; k < HID; ++k) s += b2[k] * W3[(size_t)k * DOTD + j];
  bc[j] = s;
}

__global__ __launch_bounds__(256) void k_layernorm(
    float* __restrict__ x, const float* __restrict__ gm,
    const float* __restrict__ bt, int R)
{
  int wave = threadIdx.x >> 5, lane = threadIdx.x & 31;
  int row = blockIdx.x * 8 + wave;
  if (row >= R) return;
  float* xr = x + (size_t)row * DOTD;
  float v[8]; float s = 0.f;
  #pragma unroll
  for (int i = 0; i < 8; ++i) { v[i] = xr[lane * 8 + i]; s += v[i]; }
  for (int off = 1; off < 32; off <<= 1) s += __shfl_xor(s, off, 32);
  float mu = s * (1.f / DOTD);
  float q = 0.f;
  #pragma unroll
  for (int i = 0; i < 8; ++i) { float d = v[i] - mu; q += d * d; }
  for (int off = 1; off < 32; off <<= 1) q += __shfl_xor(q, off, 32);
  float rstd = rsqrtf(q * (1.f / DOTD) + 1e-5f);
  #pragma unroll
  for (int i = 0; i < 8; ++i) {
    int c = lane * 8 + i;
    xr[c] = (v[i] - mu) * rstd * gm[c] + bt[c];
  }
}

__global__ __launch_bounds__(256) void k_score(
    const float* __restrict__ q, const float* __restrict__ r,
    const float* __restrict__ sb, float* __restrict__ out, int E)
{
  int wave = threadIdx.x >> 5, lane = threadIdx.x & 31;
  int e = blockIdx.x * 8 + wave;
  if (e >= E) return;
  const float* qe = q + (size_t)e * DOTD;
  const float* re = r + (size_t)e * DOTD;
  float s = 0.f;
  #pragma unroll
  for (int i = 0; i < 8; ++i) s += qe[lane * 8 + i] * re[lane * 8 + i];
  for (int off = 1; off < 32; off <<= 1) s += __shfl_xor(s, off, 32);
  if (lane == 0) out[e] = s * 0.0625f + sb[0];   // 1/sqrt(256)
}

// ---------------- host orchestration ----------------
extern "C" void kernel_launch(void* const* d_in, const int* in_sizes, int n_in,
                              void* d_out, int out_size, void* d_ws, size_t ws_size,
                              hipStream_t stream)
{
  const float* xq        = (const float*)d_in[0];
  const float* xt        = (const float*)d_in[1];
  const float* xa        = (const float*)d_in[2];
  const float* afeat     = (const float*)d_in[3];
  const float* qp_W      = (const float*)d_in[4];
  const float* qp_b      = (const float*)d_in[5];
  const float* tp_W      = (const float*)d_in[6];
  const float* tp_b      = (const float*)d_in[7];
  const float* ap_W      = (const float*)d_in[8];
  const float* ap_b      = (const float*)d_in[9];
  const float* afp_W1    = (const float*)d_in[10];
  const float* afp_b1    = (const float*)d_in[11];
  const float* afp_W2    = (const float*)d_in[12];
  const float* afp_b2    = (const float*)d_in[13];
  const float* conv_Wl   = (const float*)d_in[14];
  const float* conv_bl   = (const float*)d_in[15];
  const float* conv_Wr   = (const float*)d_in[16];
  const float* conv_br   = (const float*)d_in[17];
  const float* conv_att  = (const float*)d_in[18];
  const float* conv_bias = (const float*)d_in[19];
  const float* qh_W      = (const float*)d_in[20];
  const float* qh_b      = (const float*)d_in[21];
  const float* qh_g      = (const float*)d_in[22];
  const float* qh_beta   = (const float*)d_in[23];
  const float* rh_W      = (const float*)d_in[24];
  const float* rh_b      = (const float*)d_in[25];
  const float* rh_g      = (const float*)d_in[26];
  const float* rh_beta   = (const float*)d_in[27];
  const float* score_b   = (const float*)d_in[28];
  const int*   ei_gr     = (const int*)d_in[29];
  const int*   ei_lt     = (const int*)d_in[30];
  const int*   ei_sim    = (const int*)d_in[31];
  const int*   ei_comp   = (const int*)d_in[32];
  const int*   ei_gen    = (const int*)d_in[33];

  // ---- workspace arena (floats) ----
  float* ws = (float*)d_ws;
  size_t o = 0;
  float* HQ = ws + o; o += (size_t)NQ * HID;
  float* HT = ws + o; o += (size_t)NT * HID;
  float* HA = ws + o; o += (size_t)NA * HID;
  float* AQ = ws + o; o += (size_t)NQ * HID;
  float* AT = ws + o; o += (size_t)NT * HID;
  float* AA = ws + o; o += (size_t)NA * HID;
  float* XL = ws + o; o += (size_t)NT * HID;     // max src-node count
  float* XR = ws + o; o += (size_t)NT * HID;
  float* LG = ws + o; o += (size_t)E_COMP * NHD; // max(E) incl. sim+loops
  float* EX = ws + o; o += (size_t)E_COMP * NHD;
  float* LM = ws + o; o += (size_t)NT * NHD;
  float* DN = ws + o; o += (size_t)NT * NHD;
  // head-phase aliases (layer scratch dead by then)
  float* QB  = XL;               // E_GEN*256
  float* RB  = XR;               // E_GEN*256
  float* AFH = AT;               // E_GEN*64
  float* WC  = AA;               // 64*256
  float* BC  = AA + 64 * DOTD;   // 256

  auto gemm = [&](const float* A, const float* B, const float* bias, float* C,
                  const int* idx, int M, int N, int K, int flags) {
    dim3 grid(N / 64, (M + 127) / 128);
    gemm_wmma_f16<<<grid, dim3(256), 0, stream>>>(A, B, bias, C, idx, M, N, K, flags);
  };
  auto grd = [](size_t n) { return (unsigned)((n + 255) / 256); };

  // ---- input projections (WMMA) ----
  gemm(xq, qp_W, qp_b, HQ, nullptr, NQ, HID, 384, FLAG_BIAS);
  gemm(xt, tp_W, tp_b, HT, nullptr, NT, HID, 384, FLAG_BIAS);
  gemm(xa, ap_W, ap_b, HA, nullptr, NA, HID, 384, FLAG_BIAS);

  // ---- relation table ----
  struct Rel { const int* src; const int* dst; const float* xs; const float* xd;
               float* acc; int ns, nd, E, nloop; };
  Rel rel[6] = {
    { ei_gr,          ei_gr + E_GR,     HQ, HT, AT, NQ, NT, E_GR,   0  },  // q->t
    { ei_gr + E_GR,   ei_gr,            HT, HQ, AQ, NT, NQ, E_GR,   0  },  // t->q
    { ei_lt,          ei_lt + E_LT,     HT, HA, AA, NT, NA, E_LT,   0  },  // t->a
    { ei_lt + E_LT,   ei_lt,            HA, HT, AT, NA, NT, E_LT,   0  },  // a->t
    { ei_sim,         ei_sim + E_SIM,   HQ, HQ, AQ, NQ, NQ, E_SIM,  NQ },  // q->q (+loops)
    { ei_comp,        ei_comp + E_COMP, HT, HT, AT, NT, NT, E_COMP, 0  },  // t->t
  };
  auto cb = [&](int l, int r) { return conv_bias + ((size_t)(l * 6 + r)) * HID; };

  for (int l = 0; l < NL; ++l) {
    // seed relation-mean accumulators with summed per-relation biases
    k_acc_init<<<grd((size_t)NQ*HID), 256, 0, stream>>>(AQ, cb(l,1), cb(l,4), nullptr, (size_t)NQ*HID);
    k_acc_init<<<grd((size_t)NT*HID), 256, 0, stream>>>(AT, cb(l,0), cb(l,3), cb(l,5), (size_t)NT*HID);
    k_acc_init<<<grd((size_t)NA*HID), 256, 0, stream>>>(AA, cb(l,2), nullptr, nullptr, (size_t)NA*HID);

    for (int r = 0; r < 6; ++r) {
      const Rel& R = rel[r];
      const float* Wl  = conv_Wl  + ((size_t)(l * 6 + r)) * HID * HID;
      const float* bl  = conv_bl  + ((size_t)(l * 6 + r)) * HID;
      const float* Wr  = conv_Wr  + ((size_t)(l * 6 + r)) * HID * HID;
      const float* br  = conv_br  + ((size_t)(l * 6 + r)) * HID;
      const float* att = conv_att + ((size_t)(l * 6 + r)) * NHD * HC;

      gemm(R.xs, Wl, bl, XL, nullptr, R.ns, HID, HID, FLAG_BIAS);   // lin_l(x_src)
      gemm(R.xd, Wr, br, XR, nullptr, R.nd, HID, HID, FLAG_BIAS);   // lin_r(x_dst)

      k_fill<<<grd((size_t)R.nd * NHD), 256, 0, stream>>>(LM, (size_t)R.nd * NHD, -INFINITY);
      k_fill<<<grd((size_t)R.nd * NHD), 256, 0, stream>>>(DN, (size_t)R.nd * NHD, 0.f);

      k_edge_logits<<<(R.E + 7) / 8, 256, 0, stream>>>(R.src, R.dst, R.E, 0, XL, XR, att, LG, LM, 0);
      if (R.nloop)
        k_edge_logits<<<(R.nloop + 7) / 8, 256, 0, stream>>>(nullptr, nullptr, R.nloop, 1, XL, XR, att, LG, LM, R.E);

      k_edge_exp<<<grd((size_t)R.E * NHD), 256, 0, stream>>>(R.dst, R.E, 0, LG, 0, LM, EX, DN);
      if (R.nloop)
        k_edge_exp<<<grd((size_t)R.nloop * NHD), 256, 0, stream>>>(nullptr, R.nloop, 1, LG, R.E, LM, EX, DN);

      k_edge_scatter<<<(R.E + 7) / 8, 256, 0, stream>>>(R.src, R.dst, R.E, 0, XL, EX, 0, DN, R.acc);
      if (R.nloop)
        k_edge_scatter<<<(R.nloop + 7) / 8, 256, 0, stream>>>(nullptr, nullptr, R.nloop, 1, XL, EX, R.E, DN, R.acc);
    }

    // HeteroConv mean + relu
    k_finalize<<<grd((size_t)NQ*HID), 256, 0, stream>>>(HQ, AQ, (size_t)NQ*HID, 0.5f);
    k_finalize<<<grd((size_t)NT*HID), 256, 0, stream>>>(HT, AT, (size_t)NT*HID, 1.f/3.f);
    k_finalize<<<grd((size_t)NA*HID), 256, 0, stream>>>(HA, AA, (size_t)NA*HID, 1.f);
  }

  // ---- head ----
  const int* gen_src = ei_gen;
  const int* gen_dst = ei_gen + E_GEN;
  const float* rh_W3 = rh_W + (size_t)2 * HID * DOTD;  // af_emb slice of rh_W

  k_afp1<<<grd((size_t)E_GEN * 64), 256, 0, stream>>>(afeat, afp_W1, afp_b1, AFH, E_GEN);
  // pre-contract afp_W2 @ rh_W[1024:] -> [64,256] so af_emb never materializes
  gemm(afp_W2, rh_W3, nullptr, WC, nullptr, 64, DOTD, HID, 0);
  k_bc<<<1, 256, 0, stream>>>(afp_b2, rh_W3, rh_b, BC);

  // q head: gather hq[gen_src] folded into GEMM
  gemm(HQ, qh_W, qh_b, QB, gen_src, E_GEN, DOTD, HID, FLAG_BIAS);
  // r head: concat folded into 3 accumulating GEMMs
  gemm(HT, rh_W,                     BC,      RB, gen_dst, E_GEN, DOTD, HID, FLAG_BIAS);
  gemm(HA, rh_W + (size_t)HID*DOTD,  nullptr, RB, gen_dst, E_GEN, DOTD, HID, FLAG_ACC);
  gemm(AFH, WC,                      nullptr, RB, nullptr, E_GEN, DOTD, 64,  FLAG_ACC);

  k_layernorm<<<(E_GEN + 7) / 8, 256, 0, stream>>>(QB, qh_g, qh_beta, E_GEN);
  k_layernorm<<<(E_GEN + 7) / 8, 256, 0, stream>>>(RB, rh_g, rh_beta, E_GEN);
  k_score<<<(E_GEN + 7) / 8, 256, 0, stream>>>(QB, RB, score_b, (float*)d_out, E_GEN);
}
